// GPT2Model_47502338294349
// MI455X (gfx1250) — compile-verified, matrix operand order checked by reading, not run on hardware
//
#include <hip/hip_runtime.h>
#include <hip/hip_bf16.h>

typedef __bf16 bf16_t;
typedef bf16_t bf16x4  __attribute__((ext_vector_type(4)));
typedef bf16_t bf16x8  __attribute__((ext_vector_type(8)));
typedef bf16_t bf16x16 __attribute__((ext_vector_type(16)));
typedef float  f32x8   __attribute__((ext_vector_type(8)));

#define E_DIM   1024
#define HEADS   16
#define DHEAD   64
#define LAYERS  4
#define SEQ     1024
#define BATCH   4
#define MROWS   (BATCH*SEQ)   /* 4096 */

// ---------------------------------------------------------------------------
// WMMA helper: D = A(16x32 bf16) * B(32x16 bf16) + C(16x16 f32)
// ---------------------------------------------------------------------------
__device__ __forceinline__ f32x8 wmma_bf16(bf16x16 a, bf16x16 b, f32x8 c) {
  return __builtin_amdgcn_wmma_f32_16x16x32_bf16(
      /*neg_a=*/false, a, /*neg_b=*/false, b,
      /*c_mod=*/(short)0, c, /*reuse_a=*/false, /*reuse_b=*/false);
}

// Load one 16-element bf16 operand fragment from an LDS row.
// ISA 16-bit A/B layout: lane holds row/col (lane%16); elements 0..7 <-
// K = half*8 + 0..7, elements 8..15 <- K = 16 + half*8 + 0..7 (half = lane/16).
__device__ __forceinline__ bf16x16 load_frag(const bf16_t* base, int half) {
  union { bf16x16 v; bf16x8 h[2]; } u;
  u.h[0] = *(const bf16x8*)(base + half * 8);
  u.h[1] = *(const bf16x8*)(base + 16 + half * 8);
  return u.v;
}

// ---------------------------------------------------------------------------
// Embeddings: h[b,s,:] = wte[ids[b,s],:] + wpe[s,:]
// ---------------------------------------------------------------------------
__global__ __launch_bounds__(256) void embed_kernel(
    const int* __restrict__ ids, const float* __restrict__ wte,
    const float* __restrict__ wpe, float* __restrict__ h) {
  int i = blockIdx.x * 256 + threadIdx.x;     // < MROWS*E
  int e = i & (E_DIM - 1);
  int m = i >> 10;
  int s = m & (SEQ - 1);
  h[i] = wte[(size_t)ids[m] * E_DIM + e] + wpe[(size_t)s * E_DIM + e];
}

// ---------------------------------------------------------------------------
// LayerNorm over last dim (E=1024), one block (256 thr) per row
// ---------------------------------------------------------------------------
__global__ __launch_bounds__(256) void ln_kernel(
    const float* __restrict__ in, const float* __restrict__ g,
    const float* __restrict__ bb, float* __restrict__ out) {
  __shared__ float red[256];
  int row = blockIdx.x;
  int t = threadIdx.x;
  const float* p = in + (size_t)row * E_DIM;
  float4 v = *(const float4*)(p + t * 4);

  red[t] = v.x + v.y + v.z + v.w;
  __syncthreads();
  for (int o = 128; o; o >>= 1) { if (t < o) red[t] += red[t + o]; __syncthreads(); }
  float mean = red[0] * (1.0f / E_DIM);
  __syncthreads();

  float dx = v.x - mean, dy = v.y - mean, dz = v.z - mean, dw = v.w - mean;
  red[t] = dx * dx + dy * dy + dz * dz + dw * dw;
  __syncthreads();
  for (int o = 128; o; o >>= 1) { if (t < o) red[t] += red[t + o]; __syncthreads(); }
  float rstd = rsqrtf(red[0] * (1.0f / E_DIM) + 1e-5f);

  float4 gg = *(const float4*)(g + t * 4);
  float4 bv = *(const float4*)(bb + t * 4);
  float4 o4;
  o4.x = dx * rstd * gg.x + bv.x;
  o4.y = dy * rstd * gg.y + bv.y;
  o4.z = dz * rstd * gg.z + bv.z;
  o4.w = dw * rstd * gg.w + bv.w;
  *(float4*)(out + (size_t)row * E_DIM + t * 4) = o4;
}

// ---------------------------------------------------------------------------
// Presents: kt[b,h,d,s] = k ; v_out[b,h,s,d] = v   (from this layer's qkv)
// ---------------------------------------------------------------------------
__global__ __launch_bounds__(256) void presents_kernel(
    const float* __restrict__ qkv, float* __restrict__ kt_out,
    float* __restrict__ v_out) {
  int i = blockIdx.x * 256 + threadIdx.x;     // < 4*16*1024*64
  int d  = i & 63;
  int s  = (i >> 6) & (SEQ - 1);
  int hh = (i >> 16) & (HEADS - 1);
  int b  = i >> 20;
  const float* base = qkv + (size_t)(b * SEQ + s) * (3 * E_DIM) + hh * DHEAD + d;
  kt_out[(((size_t)(b * HEADS + hh) * DHEAD + d) << 10) + s] = base[E_DIM];
  v_out[i] = base[2 * E_DIM];
}

// ---------------------------------------------------------------------------
// Tiled bf16-WMMA GEMM with register-pipelined staging.
// Block tile 128x128, 8 waves (2M x 4N), wave tile 64x32, K chunk 32.
// A fp32 [M,K] row-major, W fp32 [K,N] row-major; converted to bf16 in LDS.
// ---------------------------------------------------------------------------
__device__ __forceinline__ void gemm_load_regs(
    const float* __restrict__ A, const float* __restrict__ W,
    int m0, int n0, int N, int K, int k0, int t,
    float4 (&ra)[4], float4 (&rb)[4]) {
#pragma unroll
  for (int i = 0; i < 4; ++i) {                 // A tile: 128 rows x 32 k
    int s = t + i * 256;
    int row = s >> 3, kk = (s & 7) << 2;
    ra[i] = *(const float4*)(A + (size_t)(m0 + row) * K + k0 + kk);
  }
#pragma unroll
  for (int i = 0; i < 4; ++i) {                 // W tile: 32 k x 128 n
    int s = t + i * 256;
    int kr = s >> 5, ng = (s & 31) << 2;
    rb[i] = *(const float4*)(W + (size_t)(k0 + kr) * N + n0 + ng);
  }
}

__device__ __forceinline__ void gemm_store_lds(
    bf16_t (*__restrict__ lA)[40], bf16_t (*__restrict__ lB)[40], int t,
    const float4 (&ra)[4], const float4 (&rb)[4]) {
#pragma unroll
  for (int i = 0; i < 4; ++i) {
    int s = t + i * 256;
    int row = s >> 3, kk = (s & 7) << 2;
    bf16x4 p = {(bf16_t)ra[i].x, (bf16_t)ra[i].y, (bf16_t)ra[i].z, (bf16_t)ra[i].w};
    *(bf16x4*)&lA[row][kk] = p;
  }
#pragma unroll
  for (int i = 0; i < 4; ++i) {                 // transposed: lB[n][k]
    int s = t + i * 256;
    int kr = s >> 5, ng = (s & 31) << 2;
    lB[ng + 0][kr] = (bf16_t)rb[i].x;
    lB[ng + 1][kr] = (bf16_t)rb[i].y;
    lB[ng + 2][kr] = (bf16_t)rb[i].z;
    lB[ng + 3][kr] = (bf16_t)rb[i].w;
  }
}

__global__ __launch_bounds__(256) void gemm_kernel(
    const float* __restrict__ A, const float* __restrict__ W,
    const float* __restrict__ bias, const float* __restrict__ abias,
    const float* __restrict__ res, float* __restrict__ C,
    int N, int K, int gelu_flag) {
  __shared__ __align__(16) bf16_t lA[2][128][40];  // [m][k], padded
  __shared__ __align__(16) bf16_t lB[2][128][40];  // [n][k], transposed, padded
  int t = threadIdx.x;
  int lane = t & 31, w = t >> 5;
  int wm = w & 1, wn = w >> 1;
  int l16 = lane & 15, lh = lane >> 4;
  int m0 = blockIdx.y * 128;
  int n0 = blockIdx.x * 128;
  int nk = K >> 5;

  f32x8 acc[4][2] = {};
  float4 ra[4], rb[4];

  gemm_load_regs(A, W, m0, n0, N, K, 0, t, ra, rb);
  gemm_store_lds(lA[0], lB[0], t, ra, rb);
  __syncthreads();

  for (int kt = 0; kt < nk; ++kt) {
    int cur = kt & 1;
    bool has_next = (kt + 1 < nk);
    if (has_next)                                // issue global loads early
      gemm_load_regs(A, W, m0, n0, N, K, (kt + 1) << 5, t, ra, rb);

    bf16x16 aF[4], bF[2];
#pragma unroll
    for (int i = 0; i < 4; ++i)
      aF[i] = load_frag(&lA[cur][wm * 64 + i * 16 + l16][0], lh);
#pragma unroll
    for (int j = 0; j < 2; ++j)
      bF[j] = load_frag(&lB[cur][wn * 32 + j * 16 + l16][0], lh);
#pragma unroll
    for (int i = 0; i < 4; ++i)
#pragma unroll
      for (int j = 0; j < 2; ++j)
        acc[i][j] = wmma_bf16(aF[i], bF[j], acc[i][j]);

    if (has_next)                                // wait lands here, after WMMAs
      gemm_store_lds(lA[cur ^ 1], lB[cur ^ 1], t, ra, rb);
    __syncthreads();
  }

  // epilogue: bias (+attn_bias) (+gelu) (+residual)
#pragma unroll
  for (int i = 0; i < 4; ++i)
    for (int j = 0; j < 2; ++j)
#pragma unroll
      for (int r = 0; r < 8; ++r) {
        int row = m0 + wm * 64 + i * 16 + r + lh * 8;
        int col = n0 + wn * 32 + j * 16 + l16;
        float v = acc[i][j][r] + bias[col];
        if (abias) v += abias[(size_t)(row & (SEQ - 1)) * N + col];
        if (gelu_flag) v = 0.5f * v * (1.0f + erff(v * 0.70710678118f));
        if (res) v += res[(size_t)row * N + col];
        C[(size_t)row * N + col] = v;
      }
}

// ---------------------------------------------------------------------------
// Flash attention (no causal mask). One block per (b, h, 128 q-rows).
// 8 waves, each owns a 16-row q tile; K/V double-buffered 32 rows at a time.
// ---------------------------------------------------------------------------
__device__ __forceinline__ void attn_load_regs(
    const float* __restrict__ qkv, int b, int h, int kt0, int t,
    float4 (&rk)[2], float4 (&rv)[2]) {
#pragma unroll
  for (int i = 0; i < 2; ++i) {
    int s = t + i * 256;
    int kr = s >> 4, dg = (s & 15) << 2;
    const float* kp = qkv + (size_t)(b * SEQ + kt0 + kr) * (3 * E_DIM)
                      + E_DIM + h * DHEAD + dg;
    rk[i] = *(const float4*)kp;
    rv[i] = *(const float4*)(kp + E_DIM);
  }
}

__device__ __forceinline__ void attn_store_lds(
    bf16_t (*__restrict__ lk)[72], bf16_t (*__restrict__ lv)[40], int t,
    const float4 (&rk)[2], const float4 (&rv)[2]) {
#pragma unroll
  for (int i = 0; i < 2; ++i) {
    int s = t + i * 256;
    int kr = s >> 4, dg = (s & 15) << 2;
    bf16x4 pk = {(bf16_t)rk[i].x, (bf16_t)rk[i].y, (bf16_t)rk[i].z, (bf16_t)rk[i].w};
    *(bf16x4*)&lk[kr][dg] = pk;
    lv[dg + 0][kr] = (bf16_t)rv[i].x;            // transposed: lv[d][krow]
    lv[dg + 1][kr] = (bf16_t)rv[i].y;
    lv[dg + 2][kr] = (bf16_t)rv[i].z;
    lv[dg + 3][kr] = (bf16_t)rv[i].w;
  }
}

__global__ __launch_bounds__(256) void attn_kernel(
    const float* __restrict__ qkv, float* __restrict__ out) {
  __shared__ __align__(16) bf16_t lK[2][32][72];   // [krow][d]
  __shared__ __align__(16) bf16_t lV[2][64][40];   // [d][krow]
  __shared__ __align__(16) bf16_t lP[8][16][40];   // per-wave P scratch
  int t = threadIdx.x;
  int lane = t & 31, w = t >> 5;
  int l16 = lane & 15, lh = lane >> 4;
  int qb = blockIdx.x & 7;
  int h  = (blockIdx.x >> 3) & (HEADS - 1);
  int b  = blockIdx.x >> 7;

  // per-wave Q fragments (16 rows x 64 D), scaled by 1/sqrt(64)
  bf16x16 qA[2];
  {
    int qrow = qb * 128 + w * 16 + l16;
    const float* qp = qkv + (size_t)(b * SEQ + qrow) * (3 * E_DIM) + h * DHEAD;
#pragma unroll
    for (int c = 0; c < 2; ++c) {
      union { bf16x16 v; bf16_t e[16]; } u;
#pragma unroll
      for (int g = 0; g < 2; ++g) {
        int d0 = c * 32 + g * 16 + lh * 8;
        float4 f0 = *(const float4*)(qp + d0);
        float4 f1 = *(const float4*)(qp + d0 + 4);
        u.e[g * 8 + 0] = (bf16_t)(f0.x * 0.125f);
        u.e[g * 8 + 1] = (bf16_t)(f0.y * 0.125f);
        u.e[g * 8 + 2] = (bf16_t)(f0.z * 0.125f);
        u.e[g * 8 + 3] = (bf16_t)(f0.w * 0.125f);
        u.e[g * 8 + 4] = (bf16_t)(f1.x * 0.125f);
        u.e[g * 8 + 5] = (bf16_t)(f1.y * 0.125f);
        u.e[g * 8 + 6] = (bf16_t)(f1.z * 0.125f);
        u.e[g * 8 + 7] = (bf16_t)(f1.w * 0.125f);
      }
      qA[c] = u.v;
    }
  }

  float m_i[8], l_i[8];
#pragma unroll
  for (int r = 0; r < 8; ++r) { m_i[r] = -1e30f; l_i[r] = 0.0f; }
  f32x8 acc[4] = {};

  float4 rk[2], rv[2];
  attn_load_regs(qkv, b, h, 0, t, rk, rv);
  attn_store_lds(lK[0], lV[0], t, rk, rv);
  __syncthreads();

  for (int it = 0; it < SEQ / 32; ++it) {
    int cur = it & 1;
    bool has_next = (it + 1 < SEQ / 32);
    if (has_next)                                 // prefetch next K/V tile
      attn_load_regs(qkv, b, h, (it + 1) * 32, t, rk, rv);

    // scores: S(16x32) = Q(16x64) @ K^T, two 16-col halves
    f32x8 S0 = {}, S1 = {};
#pragma unroll
    for (int c = 0; c < 2; ++c) {
      bf16x16 k0f = load_frag(&lK[cur][l16][c * 32], lh);
      bf16x16 k1f = load_frag(&lK[cur][16 + l16][c * 32], lh);
      S0 = wmma_bf16(qA[c], k0f, S0);
      S1 = wmma_bf16(qA[c], k1f, S1);
    }

    // online softmax (row stats live per lane for 8 rows each)
    float mx[8], rs[8], corr[8];
#pragma unroll
    for (int r = 0; r < 8; ++r) mx[r] = fmaxf(S0[r], S1[r]);
#pragma unroll
    for (int off = 8; off >= 1; off >>= 1)
#pragma unroll
      for (int r = 0; r < 8; ++r) mx[r] = fmaxf(mx[r], __shfl_xor(mx[r], off, 32));
#pragma unroll
    for (int r = 0; r < 8; ++r) {
      float mn = fmaxf(m_i[r], mx[r]);
      corr[r] = __expf(m_i[r] - mn);
      m_i[r] = mn;
      S0[r] = __expf(S0[r] - mn);
      S1[r] = __expf(S1[r] - mn);
      rs[r] = S0[r] + S1[r];
    }
#pragma unroll
    for (int off = 8; off >= 1; off >>= 1)
#pragma unroll
      for (int r = 0; r < 8; ++r) rs[r] += __shfl_xor(rs[r], off, 32);
#pragma unroll
    for (int r = 0; r < 8; ++r) l_i[r] = l_i[r] * corr[r] + rs[r];
#pragma unroll
    for (int tt = 0; tt < 4; ++tt)
#pragma unroll
      for (int r = 0; r < 8; ++r) acc[tt][r] *= corr[r];

    // re-layout P: C-layout f32 -> bf16 A-layout via per-wave LDS scratch
#pragma unroll
    for (int r = 0; r < 8; ++r) {
      int row = r + lh * 8;
      lP[w][row][l16]      = (bf16_t)S0[r];
      lP[w][row][16 + l16] = (bf16_t)S1[r];
    }
    asm volatile("s_wait_dscnt 0" ::: "memory");

    bf16x16 pF = load_frag(&lP[w][l16][0], lh);
#pragma unroll
    for (int tt = 0; tt < 4; ++tt) {
      bf16x16 vF = load_frag(&lV[cur][tt * 16 + l16][0], lh);
      acc[tt] = wmma_bf16(pF, vF, acc[tt]);
    }

    if (has_next)                                 // store prefetched tile
      attn_store_lds(lK[cur ^ 1], lV[cur ^ 1], t, rk, rv);
    __syncthreads();
  }

  // write O = acc / l  into attn_out [B*S, E] at head columns
#pragma unroll
  for (int tt = 0; tt < 4; ++tt)
#pragma unroll
    for (int r = 0; r < 8; ++r) {
      int row = qb * 128 + w * 16 + r + lh * 8;
      int col = h * DHEAD + tt * 16 + l16;
      out[(size_t)(b * SEQ + row) * E_DIM + col] = acc[tt][r] / l_i[r];
    }
}

// ---------------------------------------------------------------------------
extern "C" void kernel_launch(void* const* d_in, const int* in_sizes, int n_in,
                              void* d_out, int out_size, void* d_ws, size_t ws_size,
                              hipStream_t stream) {
  const int*   ids     = (const int*)  d_in[0];
  const float* wte     = (const float*)d_in[1];
  const float* wpe     = (const float*)d_in[2];
  const float* ln1_g   = (const float*)d_in[3];
  const float* ln1_b   = (const float*)d_in[4];
  const float* w_attn  = (const float*)d_in[5];
  const float* b_attn  = (const float*)d_in[6];
  const float* attn_bias = (const float*)d_in[7];
  const float* w_cproj = (const float*)d_in[8];
  const float* b_cproj = (const float*)d_in[9];
  const float* ln2_g   = (const float*)d_in[10];
  const float* ln2_b   = (const float*)d_in[11];
  const float* w_fc    = (const float*)d_in[12];
  const float* b_fc    = (const float*)d_in[13];
  const float* w_mproj = (const float*)d_in[14];
  const float* b_mproj = (const float*)d_in[15];
  const float* lnf_g   = (const float*)d_in[16];
  const float* lnf_b   = (const float*)d_in[17];

  float* out = (float*)d_out;
  float* ws  = (float*)d_ws;

  // workspace layout (floats): h | x | attn_out | mlp | qkv  (~160 MB total)
  float* h        = ws;
  float* x        = h + (size_t)MROWS * E_DIM;
  float* attn_out = x + (size_t)MROWS * E_DIM;
  float* mlp      = attn_out + (size_t)MROWS * E_DIM;
  float* qkv      = mlp + (size_t)MROWS * 4 * E_DIM;

  dim3 blk(256);

  embed_kernel<<<(MROWS * E_DIM) / 256, blk, 0, stream>>>(ids, wte, wpe, h);

  for (int l = 0; l < LAYERS; ++l) {
    const float* wA = w_attn  + (size_t)l * E_DIM * 3 * E_DIM;
    const float* bA = b_attn  + (size_t)l * 3 * E_DIM;
    const float* aB = attn_bias + (size_t)l * SEQ * E_DIM;
    const float* wC = w_cproj + (size_t)l * E_DIM * E_DIM;
    const float* bC = b_cproj + (size_t)l * E_DIM;
    const float* wF = w_fc    + (size_t)l * E_DIM * 4 * E_DIM;
    const float* bF = b_fc    + (size_t)l * 4 * E_DIM;
    const float* wM = w_mproj + (size_t)l * 4 * E_DIM * E_DIM;
    const float* bM = b_mproj + (size_t)l * E_DIM;

    ln_kernel<<<MROWS, blk, 0, stream>>>(h, ln1_g + l * E_DIM, ln1_b + l * E_DIM, x);

    gemm_kernel<<<dim3(3 * E_DIM / 128, MROWS / 128), blk, 0, stream>>>(
        x, wA, bA, nullptr, nullptr, qkv, 3 * E_DIM, E_DIM, 0);

    float* kt_out = out + (size_t)MROWS * E_DIM + (size_t)l * 2 * MROWS * E_DIM;
    float* v_out  = kt_out + (size_t)MROWS * E_DIM;
    presents_kernel<<<(MROWS * E_DIM) / 256, blk, 0, stream>>>(qkv, kt_out, v_out);

    attn_kernel<<<BATCH * HEADS * (SEQ / 128), blk, 0, stream>>>(qkv, attn_out);

    gemm_kernel<<<dim3(E_DIM / 128, MROWS / 128), blk, 0, stream>>>(
        attn_out, wC, bC, aB, h, h, E_DIM, E_DIM, 0);

    ln_kernel<<<MROWS, blk, 0, stream>>>(h, ln2_g + l * E_DIM, ln2_b + l * E_DIM, x);

    gemm_kernel<<<dim3(4 * E_DIM / 128, MROWS / 128), blk, 0, stream>>>(
        x, wF, bF, nullptr, nullptr, mlp, 4 * E_DIM, E_DIM, 1);

    gemm_kernel<<<dim3(E_DIM / 128, MROWS / 128), blk, 0, stream>>>(
        mlp, wM, bM, nullptr, h, h, E_DIM, 4 * E_DIM, 0);
  }

  ln_kernel<<<MROWS, blk, 0, stream>>>(h, lnf_g, lnf_b, out);
}
